// _SelfAttentionBlock_73323681677920
// MI455X (gfx1250) — matrix operand order repaired and checked by reading, not compiled, not run on hardware
//
#include <hip/hip_runtime.h>
#include <hip/hip_bf16.h>

typedef __attribute__((ext_vector_type(16))) __bf16 bf16x16;
typedef __attribute__((ext_vector_type(8)))  __bf16 bf16x8;
typedef __attribute__((ext_vector_type(8)))  float  f32x8;

#ifndef __has_builtin
#define __has_builtin(x) 0
#endif
#if __has_builtin(__builtin_amdgcn_global_load_async_to_lds_b128)
#define HAVE_ASYNC_LDS 1
#else
#define HAVE_ASYNC_LDS 0
#endif

#if HAVE_ASYNC_LDS
typedef int v4i_vs __attribute__((vector_size(16)));
typedef __attribute__((address_space(1))) v4i_vs gv4i;
typedef __attribute__((address_space(3))) v4i_vs lv4i;
__device__ __forceinline__ void async_b128(const void* g, void* l) {
  __builtin_amdgcn_global_load_async_to_lds_b128((gv4i*)g, (lv4i*)l, 0, 0);
}
__device__ __forceinline__ void async_wait0() {
#if __has_builtin(__builtin_amdgcn_s_wait_asynccnt)
  __builtin_amdgcn_s_wait_asynccnt(0);
#else
  asm volatile("s_wait_asynccnt 0x0" ::: "memory");
#endif
}
#endif

// ---------------------------------------------------------------------------
// WMMA fragment loaders (CDNA5 16x16x32 bf16 layouts, wave32)
// A (16M x 32K), row-major, stride ld:
//   lane l: m = l&15, k0 = (l>>4)*8 ; VGPR0-3 = K[k0..k0+7], VGPR4-7 = K[k0+16..k0+23]
// B (32K x 16N), stored column(N)-major (each column j = 32 contiguous K elems):
//   lane l: j = l&15, kb = (l>>4)*8 ; VGPR0-3 = K[kb..kb+7], VGPR4-7 = K[kb+16..kb+23]
// ---------------------------------------------------------------------------
__device__ __forceinline__ bf16x16 load_a_frag(const __bf16* base, int ld) {
  int lane = threadIdx.x & 31;
  int m    = lane & 15;
  int k0   = (lane >> 4) << 3;
  const __bf16* p = base + m * ld + k0;
  bf16x8 lo = *(const bf16x8*)(p);
  bf16x8 hi = *(const bf16x8*)(p + 16);
  bf16x16 f;
#pragma unroll
  for (int i = 0; i < 8; ++i) { f[i] = lo[i]; f[i + 8] = hi[i]; }
  return f;
}

__device__ __forceinline__ bf16x16 load_b_frag(const __bf16* base, int ld) {
  int lane = threadIdx.x & 31;
  int j    = lane & 15;
  int kb   = (lane >> 4) << 3;
  const __bf16* p = base + j * ld + kb;
  bf16x8 lo = *(const bf16x8*)(p);
  bf16x8 hi = *(const bf16x8*)(p + 16);
  bf16x16 f;
#pragma unroll
  for (int i = 0; i < 8; ++i) { f[i] = lo[i]; f[i + 8] = hi[i]; }
  return f;
}

__device__ __forceinline__ f32x8 wmma_bf16(bf16x16 a, bf16x16 b, f32x8 c) {
  return __builtin_amdgcn_wmma_f32_16x16x32_bf16(false, a, false, b, (short)0, c,
                                                 false, false);
}

__device__ __forceinline__ float half16_max(float v) {
  v = fmaxf(v, __shfl_xor(v, 1, 32));
  v = fmaxf(v, __shfl_xor(v, 2, 32));
  v = fmaxf(v, __shfl_xor(v, 4, 32));
  v = fmaxf(v, __shfl_xor(v, 8, 32));
  return v;
}
__device__ __forceinline__ float half16_sum(float v) {
  v += __shfl_xor(v, 1, 32);
  v += __shfl_xor(v, 2, 32);
  v += __shfl_xor(v, 4, 32);
  v += __shfl_xor(v, 8, 32);
  return v;
}

// ---------------------------------------------------------------------------
// small prep kernels
// ---------------------------------------------------------------------------
__global__ void f32_to_bf16_kernel(const float* __restrict__ s, __bf16* __restrict__ d, int n) {
  int i = blockIdx.x * blockDim.x + threadIdx.x;
  if (i < n) d[i] = (__bf16)s[i];
}

__global__ void affine_prep_kernel(const float* __restrict__ g, const float* __restrict__ beta,
                                   const float* __restrict__ m, const float* __restrict__ v,
                                   const float* __restrict__ bias,
                                   float* __restrict__ scale, float* __restrict__ shift,
                                   int n, int has_bn) {
  int i = blockIdx.x * blockDim.x + threadIdx.x;
  if (i >= n) return;
  if (has_bn) {
    float s = g[i] * rsqrtf(v[i] + 1e-5f);
    scale[i] = s;
    shift[i] = beta[i] - m[i] * s + bias[i] * s;
  } else {
    scale[i] = 1.0f;
    shift[i] = bias[i];
  }
}

// x (B,C,Npix) f32 -> xT (B,Npix,C) bf16
__global__ void transpose_f32_to_bf16(const float* __restrict__ src, __bf16* __restrict__ dst,
                                      int C, int Npix) {
  __shared__ float tile[32][33];
  int b = blockIdx.z;
  const float* s = src + (long)b * C * Npix;
  __bf16* d = dst + (long)b * C * Npix;
  int c0 = blockIdx.y * 32, n0 = blockIdx.x * 32;
  int tx = threadIdx.x & 31, ty = threadIdx.x >> 5;
#pragma unroll
  for (int i = 0; i < 32; i += 8)
    tile[ty + i][tx] = s[(long)(c0 + ty + i) * Npix + n0 + tx];
  __syncthreads();
#pragma unroll
  for (int i = 0; i < 32; i += 8)
    d[(long)(n0 + ty + i) * C + c0 + tx] = (__bf16)tile[tx][ty + i];
}

// ---------------------------------------------------------------------------
// conv1x1 as GEMM:  Y[n][co] = act( sum_c W[co][c] * X[n][c] * scale[co] + shift[co] )
// block tile 64(M=co) x 128(N=pix), K-step 32, 8 waves (2x4), 32x32 per wave.
// Double-buffered LDS with async global->LDS copies when available.
// OUT_MODE 0: bf16 out, pixel-major [n][co] ; OUT_MODE 1: f32 out, channel-major [co][n]
// ---------------------------------------------------------------------------
#define GT_M 64
#define GT_N 128
#define GT_K 32

template <int RELU, int OUT_MODE>
__global__ __launch_bounds__(256) void gemm_conv_kernel(
    const __bf16* __restrict__ W, const __bf16* __restrict__ X, void* __restrict__ out,
    const float* __restrict__ scale, const float* __restrict__ shift,
    int Co, int Ci, int Npix, long strideXb, long strideOb) {
  __shared__ __attribute__((aligned(16))) __bf16 As[2][GT_M][GT_K];
  __shared__ __attribute__((aligned(16))) __bf16 Bs[2][GT_N][GT_K];

  int b  = blockIdx.z;
  const __bf16* Xb = X + (long)b * strideXb;
  int n0 = blockIdx.x * GT_N;
  int m0 = blockIdx.y * GT_M;
  int tid  = threadIdx.x;
  int wave = tid >> 5;
  int wm = (wave >> 2) * 32;
  int wn = (wave & 3) * 32;

  f32x8 acc[2][2] = {};

  int ar = tid >> 2, ac = (tid & 3) * 8;   // A: 64 rows x 4 chunks(8)
  int br = tid >> 1, bc = (tid & 1) * 16;  // B: 128 rows x 2 chunks(16)

#if HAVE_ASYNC_LDS
  auto issue_tile = [&](int buf, int c0) {
    async_b128(W + (long)(m0 + ar) * Ci + c0 + ac, &As[buf][ar][ac]);
    async_b128(Xb + (long)(n0 + br) * Ci + c0 + bc, &Bs[buf][br][bc]);
    async_b128(Xb + (long)(n0 + br) * Ci + c0 + bc + 8, &Bs[buf][br][bc + 8]);
  };
  issue_tile(0, 0);
  async_wait0();
  __syncthreads();
  int nk = Ci / GT_K;
  for (int t = 0; t < nk; ++t) {
    int cur = t & 1;
    if (t + 1 < nk) issue_tile(cur ^ 1, (t + 1) * GT_K);
    bf16x16 a0 = load_a_frag(&As[cur][wm][0], GT_K);
    bf16x16 a1 = load_a_frag(&As[cur][wm + 16][0], GT_K);
    bf16x16 b0 = load_b_frag(&Bs[cur][wn][0], GT_K);
    bf16x16 b1 = load_b_frag(&Bs[cur][wn + 16][0], GT_K);
    acc[0][0] = wmma_bf16(a0, b0, acc[0][0]);
    acc[0][1] = wmma_bf16(a0, b1, acc[0][1]);
    acc[1][0] = wmma_bf16(a1, b0, acc[1][0]);
    acc[1][1] = wmma_bf16(a1, b1, acc[1][1]);
    if (t + 1 < nk) async_wait0();
    __syncthreads();
  }
#else
  for (int c0 = 0; c0 < Ci; c0 += GT_K) {
    __syncthreads();
    *(bf16x8*)(&As[0][ar][ac])      = *(const bf16x8*)(W + (long)(m0 + ar) * Ci + c0 + ac);
    *(bf16x8*)(&Bs[0][br][bc])      = *(const bf16x8*)(Xb + (long)(n0 + br) * Ci + c0 + bc);
    *(bf16x8*)(&Bs[0][br][bc + 8])  = *(const bf16x8*)(Xb + (long)(n0 + br) * Ci + c0 + bc + 8);
    if (c0 + GT_K < Ci) {
      __builtin_prefetch(W  + (long)(m0 + ar) * Ci + c0 + GT_K + ac, 0, 1);
      __builtin_prefetch(Xb + (long)(n0 + br) * Ci + c0 + GT_K + bc, 0, 1);
    }
    __syncthreads();
    bf16x16 a0 = load_a_frag(&As[0][wm][0], GT_K);
    bf16x16 a1 = load_a_frag(&As[0][wm + 16][0], GT_K);
    bf16x16 b0 = load_b_frag(&Bs[0][wn][0], GT_K);
    bf16x16 b1 = load_b_frag(&Bs[0][wn + 16][0], GT_K);
    acc[0][0] = wmma_bf16(a0, b0, acc[0][0]);
    acc[0][1] = wmma_bf16(a0, b1, acc[0][1]);
    acc[1][0] = wmma_bf16(a1, b0, acc[1][0]);
    acc[1][1] = wmma_bf16(a1, b1, acc[1][1]);
  }
#endif

  int lane = tid & 31, col = lane & 15, hi = lane >> 4;
#pragma unroll
  for (int i = 0; i < 2; ++i)
#pragma unroll
    for (int j = 0; j < 2; ++j) {
      int co_base = m0 + wm + i * 16 + hi * 8;
      int n       = n0 + wn + j * 16 + col;
      float vals[8];
#pragma unroll
      for (int r = 0; r < 8; ++r) {
        float v = acc[i][j][r] * scale[co_base + r] + shift[co_base + r];
        if (RELU) v = fmaxf(v, 0.0f);
        vals[r] = v;
      }
      if (OUT_MODE == 0) {
        __bf16* ob = (__bf16*)out + (long)b * strideOb;
        bf16x8 o;
#pragma unroll
        for (int r = 0; r < 8; ++r) o[r] = (__bf16)vals[r];
        *(bf16x8*)(ob + (long)n * Co + co_base) = o;
      } else {
        float* of = (float*)out + (long)b * strideOb;
#pragma unroll
        for (int r = 0; r < 8; ++r) of[(long)(co_base + r) * Npix + n] = vals[r];
      }
    }
}

// ---------------------------------------------------------------------------
// Flash attention: Qt,Kt,Vt (B,N,256) bf16 pixel-major -> ctx (B,N,256) bf16
// 128 queries/block (16 per wave), 64-key chunks, online softmax, scale=1/16.
// All global traffic is async global->LDS. K double-buffered; V lands raw in
// LDS ([m][cv]) and is transposed LDS->LDS into the B-operand layout, keeping
// VGPR pressure low (no long-lived register staging).
// ---------------------------------------------------------------------------
#define FA_TQ 128
#define FA_TK 64
#define FA_C  256
// Q + 2*K + 2*V^T + Vraw + P  (bytes); independent of async availability.
#define FA_LDS_BYTES ((FA_TQ * FA_C + 2 * FA_TK * FA_C + 2 * FA_C * FA_TK + FA_TK * FA_C + 8 * 16 * FA_TK) * 2)

__global__ __launch_bounds__(256) void flash_attn_kernel(
    const __bf16* __restrict__ Qt, const __bf16* __restrict__ Kt,
    const __bf16* __restrict__ Vt, __bf16* __restrict__ ctx, int Npix) {
  extern __shared__ __bf16 smem[];
  __bf16* Qs   = smem;                              // [128][256]
  __bf16* KsB  = Qs + FA_TQ * FA_C;                 // 2 x [64][256]
  __bf16* VsB  = KsB + 2 * FA_TK * FA_C;            // 2 x [256][64] (transposed)
  __bf16* Vraw = VsB + 2 * FA_C * FA_TK;            // [64][256] async landing
  __bf16* Ps   = Vraw + FA_TK * FA_C;               // 8 waves x [16][64]

  int b   = blockIdx.y;
  long boff = (long)b * Npix * FA_C;
  int q0  = blockIdx.x * FA_TQ;
  int tid = threadIdx.x, wave = tid >> 5, lane = tid & 31;
  int col = lane & 15, hi = lane >> 4;

  f32x8 Oacc[16] = {};
  float mrow[8], lrow[8];
#pragma unroll
  for (int r = 0; r < 8; ++r) { mrow[r] = -1e30f; lrow[r] = 0.0f; }
  const float sscale = 0.0625f;  // 256^-0.5

  __bf16* Qw = Qs + (wave * 16) * FA_C;
  __bf16* Pw = Ps + (wave * 16) * FA_TK;
  int nch = Npix / FA_TK;

#if HAVE_ASYNC_LDS
  {  // prologue: async Q, K0 and raw V0
    const __bf16* gq = Qt + boff + (long)q0 * FA_C;
    for (int i = tid; i < FA_TQ * FA_C / 8; i += 256) async_b128(gq + i * 8, Qs + i * 8);
    const __bf16* gk = Kt + boff;
    const __bf16* gv = Vt + boff;
    for (int i = tid; i < FA_TK * FA_C / 8; i += 256) {
      async_b128(gk + i * 8, KsB + i * 8);
      async_b128(gv + i * 8, Vraw + i * 8);
    }
    async_wait0();
  }
  __syncthreads();
#else
  {
    const bf16x8* g = (const bf16x8*)(Qt + boff + (long)q0 * FA_C);
    bf16x8* s = (bf16x8*)Qs;
    for (int i = tid; i < FA_TQ * FA_C / 8; i += 256) s[i] = g[i];
  }
#endif

  for (int ic = 0; ic < nch; ++ic) {
    int cur = ic & 1;
#if HAVE_ASYNC_LDS
    __bf16* Ks = KsB + cur * FA_TK * FA_C;
    __bf16* Vs = VsB + cur * FA_C * FA_TK;
#else
    __bf16* Ks = KsB;
    __bf16* Vs = VsB;
    __syncthreads();
    {
      const bf16x8* g = (const bf16x8*)(Kt + boff + (long)ic * FA_TK * FA_C);
      bf16x8* s = (bf16x8*)Ks;
      for (int i = tid; i < FA_TK * FA_C / 8; i += 256) s[i] = g[i];
    }
    for (int i = tid; i < FA_TK * FA_C / 8; i += 256) {
      int m = i >> 5, cv0 = (i & 31) * 8;
      bf16x8 vv = *(const bf16x8*)(Vt + boff + (long)(ic * FA_TK + m) * FA_C + cv0);
#pragma unroll
      for (int e = 0; e < 8; ++e) Vs[(cv0 + e) * FA_TK + m] = vv[e];
    }
    __syncthreads();
#endif

    // ---- S = Q K^T (16 queries x 64 keys per wave) ----
    f32x8 Sacc[4] = {};
    for (int c0 = 0; c0 < FA_C; c0 += 32) {
      bf16x16 a = load_a_frag(Qw + c0, FA_C);
#pragma unroll
      for (int mt = 0; mt < 4; ++mt) {
        bf16x16 bf = load_b_frag(Ks + (mt * 16) * FA_C + c0, FA_C);
        Sacc[mt] = wmma_bf16(a, bf, Sacc[mt]);
      }
    }

#if HAVE_ASYNC_LDS
    {  // transpose this chunk's raw V into the B-operand layout (short-lived temps)
      for (int j = 0; j < 8; ++j) {
        int i = tid + j * 256, m = i >> 5, cv0 = (i & 31) * 8;
        bf16x8 vv = *(const bf16x8*)(Vraw + m * FA_C + cv0);
#pragma unroll
        for (int e = 0; e < 8; ++e) Vs[(cv0 + e) * FA_TK + m] = vv[e];
      }
    }
#endif

    // ---- online softmax ----
    float alphar[8];
#pragma unroll
    for (int r = 0; r < 8; ++r) {
      float v = fmaxf(fmaxf(Sacc[0][r], Sacc[1][r]), fmaxf(Sacc[2][r], Sacc[3][r]));
      float mx = half16_max(v) * sscale;
      float mn = fmaxf(mrow[r], mx);
      float al = __expf(mrow[r] - mn);
      mrow[r] = mn;
      float sum = 0.0f;
#pragma unroll
      for (int mt = 0; mt < 4; ++mt) {
        float p = __expf(Sacc[mt][r] * sscale - mn);
        sum += p;
        Pw[(r + 8 * hi) * FA_TK + mt * 16 + col] = (__bf16)p;
      }
      lrow[r] = lrow[r] * al + half16_sum(sum);
      alphar[r] = al;
    }
#pragma unroll
    for (int t = 0; t < 16; ++t)
#pragma unroll
      for (int r = 0; r < 8; ++r) Oacc[t][r] *= alphar[r];

#if HAVE_ASYNC_LDS
    __syncthreads();  // A: transposes done everywhere; Vraw free; Vs[cur] ready
    if (ic + 1 < nch) {  // issue next chunk's async K + raw V, overlapping PV GEMM
      const __bf16* gk = Kt + boff + (long)(ic + 1) * FA_TK * FA_C;
      const __bf16* gv = Vt + boff + (long)(ic + 1) * FA_TK * FA_C;
      __bf16* kd = KsB + (cur ^ 1) * FA_TK * FA_C;
      for (int i = tid; i < FA_TK * FA_C / 8; i += 256) {
        async_b128(gk + i * 8, kd + i * 8);
        async_b128(gv + i * 8, Vraw + i * 8);
      }
    }
#endif

    // ---- O += P @ V (K-dim = 64 keys) ----
    bf16x16 pa0 = load_a_frag(Pw + 0, FA_TK);
    bf16x16 pa1 = load_a_frag(Pw + 32, FA_TK);
#pragma unroll
    for (int t = 0; t < 16; ++t) {
      bf16x16 vb0 = load_b_frag(Vs + (t * 16) * FA_TK + 0, FA_TK);
      bf16x16 vb1 = load_b_frag(Vs + (t * 16) * FA_TK + 32, FA_TK);
      Oacc[t] = wmma_bf16(pa0, vb0, Oacc[t]);
      Oacc[t] = wmma_bf16(pa1, vb1, Oacc[t]);
    }

#if HAVE_ASYNC_LDS
    if (ic + 1 < nch) async_wait0();
    __syncthreads();  // B: async data visible; Ks[cur]/Vs[cur] reads done
#endif
  }

  float inv[8];
#pragma unroll
  for (int r = 0; r < 8; ++r) inv[r] = 1.0f / lrow[r];
#pragma unroll
  for (int t = 0; t < 16; ++t)
#pragma unroll
    for (int r = 0; r < 8; ++r) {
      int n = q0 + wave * 16 + r + 8 * hi;
      ctx[boff + (long)n * FA_C + t * 16 + col] = (__bf16)(Oacc[t][r] * inv[r]);
    }
}

// ---------------------------------------------------------------------------
extern "C" void kernel_launch(void* const* d_in, const int* in_sizes, int n_in,
                              void* d_out, int out_size, void* d_ws, size_t ws_size,
                              hipStream_t stream) {
  constexpr int B = 4, Cin = 512, Ck = 256, Cv = 256, Cout = 512, N = 4096;
  const float* x = (const float*)d_in[0];
  const float* kp[12]; for (int i = 0; i < 12; ++i) kp[i] = (const float*)d_in[1 + i];
  const float* qp[12]; for (int i = 0; i < 12; ++i) qp[i] = (const float*)d_in[13 + i];
  const float* v_w = (const float*)d_in[25];
  const float* v_b = (const float*)d_in[26];
  const float* o_w = (const float*)d_in[27];
  const float* o_b = (const float*)d_in[28];

  size_t off = 0;
  auto alloc = [&](size_t bytes) -> char* {
    char* p = (char*)d_ws + off;
    off += (bytes + 255) & ~(size_t)255;
    return p;
  };
  __bf16* xT   = (__bf16*)alloc((size_t)B * N * Cin * 2);
  __bf16* wk1  = (__bf16*)alloc((size_t)Ck * Cin * 2);
  __bf16* wk2  = (__bf16*)alloc((size_t)Ck * Ck * 2);
  __bf16* wq1  = (__bf16*)alloc((size_t)Ck * Cin * 2);
  __bf16* wq2  = (__bf16*)alloc((size_t)Ck * Ck * 2);
  __bf16* wv   = (__bf16*)alloc((size_t)Cv * Cin * 2);
  __bf16* wo   = (__bf16*)alloc((size_t)Cout * Cv * 2);
  float* sc[6]; float* sh[6];
  int   cn[6] = {Ck, Ck, Ck, Ck, Cv, Cout};
  for (int i = 0; i < 6; ++i) { sc[i] = (float*)alloc(cn[i] * 4); sh[i] = (float*)alloc(cn[i] * 4); }
  __bf16* tmp  = (__bf16*)alloc((size_t)B * N * Ck * 2);
  __bf16* Ktb  = (__bf16*)alloc((size_t)B * N * Ck * 2);
  __bf16* Qtb  = (__bf16*)alloc((size_t)B * N * Ck * 2);
  __bf16* Vtb  = (__bf16*)alloc((size_t)B * N * Cv * 2);
  __bf16* ctxb = (__bf16*)alloc((size_t)B * N * Cv * 2);

  auto cvt = [&](const float* s, __bf16* d, int n) {
    f32_to_bf16_kernel<<<(n + 255) / 256, 256, 0, stream>>>(s, d, n);
  };
  cvt(kp[0], wk1, Ck * Cin); cvt(kp[6], wk2, Ck * Ck);
  cvt(qp[0], wq1, Ck * Cin); cvt(qp[6], wq2, Ck * Ck);
  cvt(v_w, wv, Cv * Cin);    cvt(o_w, wo, Cout * Cv);

  affine_prep_kernel<<<1, 256, 0, stream>>>(kp[2], kp[3], kp[4], kp[5], kp[1], sc[0], sh[0], Ck, 1);
  affine_prep_kernel<<<1, 256, 0, stream>>>(kp[8], kp[9], kp[10], kp[11], kp[7], sc[1], sh[1], Ck, 1);
  affine_prep_kernel<<<1, 256, 0, stream>>>(qp[2], qp[3], qp[4], qp[5], qp[1], sc[2], sh[2], Ck, 1);
  affine_prep_kernel<<<1, 256, 0, stream>>>(qp[8], qp[9], qp[10], qp[11], qp[7], sc[3], sh[3], Ck, 1);
  affine_prep_kernel<<<1, 256, 0, stream>>>(nullptr, nullptr, nullptr, nullptr, v_b, sc[4], sh[4], Cv, 0);
  affine_prep_kernel<<<(Cout + 255) / 256, 256, 0, stream>>>(nullptr, nullptr, nullptr, nullptr, o_b, sc[5], sh[5], Cout, 0);

  transpose_f32_to_bf16<<<dim3(N / 32, Cin / 32, B), 256, 0, stream>>>(x, xT, Cin, N);

  dim3 blk(256);
  gemm_conv_kernel<1, 0><<<dim3(N / GT_N, Ck / GT_M, B), blk, 0, stream>>>(
      wk1, xT, tmp, sc[0], sh[0], Ck, Cin, N, (long)N * Cin, (long)N * Ck);
  gemm_conv_kernel<1, 0><<<dim3(N / GT_N, Ck / GT_M, B), blk, 0, stream>>>(
      wk2, tmp, Ktb, sc[1], sh[1], Ck, Ck, N, (long)N * Ck, (long)N * Ck);
  gemm_conv_kernel<1, 0><<<dim3(N / GT_N, Ck / GT_M, B), blk, 0, stream>>>(
      wq1, xT, tmp, sc[2], sh[2], Ck, Cin, N, (long)N * Cin, (long)N * Ck);
  gemm_conv_kernel<1, 0><<<dim3(N / GT_N, Ck / GT_M, B), blk, 0, stream>>>(
      wq2, tmp, Qtb, sc[3], sh[3], Ck, Ck, N, (long)N * Ck, (long)N * Ck);
  gemm_conv_kernel<0, 0><<<dim3(N / GT_N, Cv / GT_M, B), blk, 0, stream>>>(
      wv, xT, Vtb, sc[4], sh[4], Cv, Cin, N, (long)N * Cin, (long)N * Cv);

  (void)hipFuncSetAttribute(reinterpret_cast<const void*>(flash_attn_kernel),
                            hipFuncAttributeMaxDynamicSharedMemorySize, FA_LDS_BYTES);
  flash_attn_kernel<<<dim3(N / FA_TQ, B), blk, FA_LDS_BYTES, stream>>>(Qtb, Ktb, Vtb, ctxb, N);

  gemm_conv_kernel<0, 1><<<dim3(N / GT_N, Cout / GT_M, B), blk, 0, stream>>>(
      wo, ctxb, d_out, sc[5], sh[5], Cout, Cv, N, (long)N * Cv, (long)Cout * N);
}
